// EquivariantAttention_13958643712328
// MI455X (gfx1250) — compile-verified, matrix operand order checked by reading, not run on hardware
//
#include <hip/hip_runtime.h>
#include <hip/hip_bf16.h>
#include <math.h>

// ---------------------------------------------------------------------------
// Types / WMMA helpers (CDNA5 / gfx1250, wave32)
// ---------------------------------------------------------------------------
typedef _Float16 half_t;
typedef __attribute__((ext_vector_type(4)))  float    v4f;
typedef __attribute__((ext_vector_type(8)))  _Float16 v8h;
typedef __attribute__((ext_vector_type(16))) _Float16 v16h;
typedef __attribute__((ext_vector_type(8)))  float    v8f;

// Load a 16x32 f16 fragment from a row-major array (ld = leading dim, elems).
// Works for A (rows = M) and for B staged N-major (rows = N).
// Per ISA 05_wmma: lane 0-15 -> row, halves0-7 = K(kb..kb+7),
// halves8-15 = K(kb+16..kb+23), kb = 8*(lane>>4).
__device__ inline v16h load_frag(const half_t* __restrict__ base, int r0, int k0, int ld) {
    const int lane = threadIdx.x & 31;
    const int kb = (lane >> 4) << 3;
    const half_t* p = base + (size_t)(r0 + (lane & 15)) * ld + k0 + kb;
    v8h lo = *(const v8h*)(p);
    v8h hi = *(const v8h*)(p + 16);
    return __builtin_shufflevector(lo, hi, 0,1,2,3,4,5,6,7,8,9,10,11,12,13,14,15);
}

__device__ inline v8f wmma_f16(v16h a, v16h b, v8f c) {
    return __builtin_amdgcn_wmma_f32_16x16x32_f16(false, a, false, b, (short)0, c, false, false);
}

// ---------------------------------------------------------------------------
// Problem constants
// ---------------------------------------------------------------------------
#define BB    4
#define NPTS  2048
#define DIM   256
#define HH    4
#define DHD   64
#define NNB   32
#define MDIM  16
#define BN    (BB * NPTS)          // 8192

// Workspace layout (bytes)
#define OFF_IDX    ((size_t)0)                     // BN*32 int
#define OFF_NBD    ((size_t)1048576)               // BN*32 f32
#define OFF_REL    ((size_t)2097152)               // BN*32*3 f32
#define OFF_QKV    ((size_t)5242880)               // BN*768 f32
#define OFF_OUTW   ((size_t)30408704)              // BN*256 f32
#define OFF_WQKVT  ((size_t)38797312)              // 768*256 f16
#define OFF_WOUTT  ((size_t)39190528)              // 256*256 f16
#define OFF_WE1T   ((size_t)39321600)              // 256*128 f16
#define OFF_WE2T   ((size_t)39387136)              // 16*256  f16

// ---------------------------------------------------------------------------
// Kernel 1: pairwise distance + top-32 nearest neighbors per (b,i)
// ---------------------------------------------------------------------------
__global__ __launch_bounds__(256) void topk_kernel(
    const float* __restrict__ coors, int* __restrict__ idx_out,
    float* __restrict__ nbd_out, float* __restrict__ rel_out) {
    __shared__ float sd[NPTS];
    __shared__ float rv[256];
    __shared__ int   ri[256];
    const int bi = blockIdx.x;
    const int b  = bi >> 11;
    const int tid = threadIdx.x;
    const float cx = coors[(size_t)bi*3+0];
    const float cy = coors[(size_t)bi*3+1];
    const float cz = coors[(size_t)bi*3+2];
    const float* cb = coors + (size_t)b * NPTS * 3;
    for (int j = tid; j < NPTS; j += 256) {
        float dx = cx - cb[j*3+0], dy = cy - cb[j*3+1], dz = cz - cb[j*3+2];
        sd[j] = sqrtf(dx*dx + dy*dy + dz*dz);
    }
    __syncthreads();
    for (int e = 0; e < NNB; ++e) {
        float bv = 3.4e38f; int bj = 0x7fffffff;
        for (int j = tid; j < NPTS; j += 256) {
            float v = sd[j];
            if (v < bv) { bv = v; bj = j; }
        }
        rv[tid] = bv; ri[tid] = bj;
        __syncthreads();
        for (int s = 128; s > 0; s >>= 1) {
            if (tid < s) {
                float ov = rv[tid+s]; int oj = ri[tid+s];
                if (ov < rv[tid] || (ov == rv[tid] && oj < ri[tid])) { rv[tid] = ov; ri[tid] = oj; }
            }
            __syncthreads();
        }
        if (tid == 0) {
            int j = ri[0];
            idx_out[(size_t)bi*NNB + e] = j;
            nbd_out[(size_t)bi*NNB + e] = rv[0];
            rel_out[((size_t)bi*NNB + e)*3 + 0] = cx - cb[j*3+0];
            rel_out[((size_t)bi*NNB + e)*3 + 1] = cy - cb[j*3+1];
            rel_out[((size_t)bi*NNB + e)*3 + 2] = cz - cb[j*3+2];
            sd[j] = 3.4e38f;
        }
        __syncthreads();
    }
}

// ---------------------------------------------------------------------------
// Kernel 2: weight prep — transpose f32 [K][N] -> f16 [N][K] (N-major)
// ---------------------------------------------------------------------------
__global__ void transpose_to_f16(const float* __restrict__ src, half_t* __restrict__ dst,
                                 int K, int Nn) {
    int x = blockIdx.x * 256 + threadIdx.x;
    if (x < K * Nn) {
        int n = x / K, k = x - n * K;
        dst[x] = (half_t)src[(size_t)k * Nn + n];
    }
}

// ---------------------------------------------------------------------------
// Kernel 3: WMMA GEMM  C[M,N] = A[M,K](f32) * B[K,N] (+bias)
// B pre-transposed N-major f16: Bt[n*K + k]. Block tile 64x64, 8 waves as
// 2(Mpair) x 4(N); each wave: two 16x16 tiles sharing one B fragment.
// LDS double-buffered: one barrier per 32-deep K step.
// ---------------------------------------------------------------------------
template <int NCOLS, int K>
__global__ __launch_bounds__(256) void gemm_wmma(
    const float* __restrict__ A, const half_t* __restrict__ Bt,
    const float* __restrict__ bias, float* __restrict__ C, int M) {
    __shared__ __align__(16) half_t sA[2][64 * 32];
    __shared__ __align__(16) half_t sB[2][64 * 32];
    const int tid  = threadIdx.x;
    const int wave = tid >> 5;
    const int lane = tid & 31;
    const int m0 = blockIdx.x * 64;
    const int n0 = blockIdx.y * 64;
    const int mr0 = (wave & 1) * 32;        // wave's M row pair base (rows mr0, mr0+16)
    const int nt  = wave >> 1;              // N tile 0..3

    // stage one 64x32 K-chunk of A (f32->f16) and B (f16): 8 elems/thread, all 16B ops
    const int sr = tid >> 2;                // 0..63
    const int sc = (tid & 3) * 8;           // 0,8,16,24
    auto stage = [&](int buf, int k0) {
        const float* ap = A + (size_t)(m0 + sr) * K + k0 + sc;
        v4f a0 = *(const v4f*)ap;
        v4f a1 = *(const v4f*)(ap + 4);
        v8h av;
        av[0] = (half_t)a0[0]; av[1] = (half_t)a0[1];
        av[2] = (half_t)a0[2]; av[3] = (half_t)a0[3];
        av[4] = (half_t)a1[0]; av[5] = (half_t)a1[1];
        av[6] = (half_t)a1[2]; av[7] = (half_t)a1[3];
        *(v8h*)&sA[buf][sr * 32 + sc] = av;
        const half_t* bp = Bt + (size_t)(n0 + sr) * K + k0 + sc;
        *(v8h*)&sB[buf][sr * 32 + sc] = *(const v8h*)bp;
    };

    constexpr int KC = K / 32;
    v8f acc0 = {}, acc1 = {};
    stage(0, 0);
    __syncthreads();
#pragma unroll
    for (int kc = 0; kc < KC; ++kc) {
        const int cur = kc & 1;
        if (kc + 1 < KC) stage(cur ^ 1, (kc + 1) * 32);
        v16h bfr = load_frag(sB[cur], nt * 16, 0, 32);
        v16h a0  = load_frag(sA[cur], mr0, 0, 32);
        v16h a1  = load_frag(sA[cur], mr0 + 16, 0, 32);
        acc0 = wmma_f16(a0, bfr, acc0);
        acc1 = wmma_f16(a1, bfr, acc1);
        __syncthreads();
    }
    const int colg = n0 + nt * 16 + (lane & 15);
    const float bv = bias ? bias[colg] : 0.0f;
    const int rbase = m0 + mr0 + ((lane >> 4) << 3);
#pragma unroll
    for (int r = 0; r < 8; ++r)
        C[(size_t)(rbase + r) * NCOLS + colg] = acc0[r] + bv;
#pragma unroll
    for (int r = 0; r < 8; ++r)
        C[(size_t)(rbase + 16 + r) * NCOLS + colg] = acc1[r] + bv;
}

// ---------------------------------------------------------------------------
// Kernel 4: fused per-(b,i): fourier enc -> pos MLP -> per-head edge MLP
// (WMMA) -> coor/attn heads -> softmax -> attention out + coords out.
// ---------------------------------------------------------------------------
__global__ __launch_bounds__(256) void fused_edge_attn(
    const float* __restrict__ qkv,        // [BN,768]
    const int*   __restrict__ nb_idx,     // [BN,32]
    const float* __restrict__ nbd,        // [BN,32]
    const float* __restrict__ relc,       // [BN,32,3]
    const half_t* __restrict__ we1t,      // [256][128]
    const float* __restrict__ be1,
    const half_t* __restrict__ we2t,      // [16][256]
    const float* __restrict__ be2,
    const float* __restrict__ wp1, const float* __restrict__ bp1,
    const float* __restrict__ wp2, const float* __restrict__ bp2,
    const float* __restrict__ wa1, const float* __restrict__ ba1,
    const float* __restrict__ wa2, const float* __restrict__ ba2,
    const float* __restrict__ wc1, const float* __restrict__ bc1,
    const float* __restrict__ wc2, const float* __restrict__ bc2,
    const float* __restrict__ wco,
    float* __restrict__ outw,             // [BN,256] (h-major heads)
    float* __restrict__ coors_out) {      // [BN,3]
    // 16KB scratch: pos-hidden f32[32][128] -> Hid f16[32][256] -> head hiddens f32
    __shared__ __align__(16) unsigned char sScratch[32 * 256 * 2];
    __shared__ __align__(16) half_t sX[32 * 128];
    __shared__ __align__(16) float sPos[32 * 64];
    __shared__ float sM[32 * 16];
    __shared__ __align__(16) float sQ[64];
    __shared__ int   sIdx[32];
    __shared__ float sNbd[32];
    __shared__ float sRel[32 * 3];
    __shared__ float sDE[32 * 9];
    __shared__ float sCW[32], sSim[32], sAttn[32];
    __shared__ float sCAcc[3];

    const int bi   = blockIdx.x;
    const int tid  = threadIdx.x;
    const int wave = tid >> 5;
    const int lane = tid & 31;
    const int brow = bi & ~(NPTS - 1);     // b * NPTS

    if (tid < 32) { sIdx[tid] = nb_idx[(size_t)bi*32 + tid]; sNbd[tid] = nbd[(size_t)bi*32 + tid]; }
    if (tid < 96) sRel[tid] = relc[(size_t)bi*96 + tid];
    if (tid < 3)  sCAcc[tid] = 0.0f;
    __syncthreads();

    // Fourier encode: cols 0-3 sin(d/2^c), 4-7 cos(d/2^(c-4)), 8 = d
    for (int x = tid; x < 32 * 9; x += 256) {
        int e = x / 9, c = x - e * 9;
        float d = sNbd[e];
        float v;
        if (c == 8) v = d;
        else {
            float xs = d / (float)(1 << (c & 3));
            v = (c < 4) ? sinf(xs) : cosf(xs);
        }
        sDE[x] = v;
    }
    __syncthreads();

    // pos MLP layer 1: [32,9]@[9,128] + relu
    float* ph = (float*)sScratch;
    for (int x = tid; x < 32 * 128; x += 256) {
        int e = x >> 7, k = x & 127;
        float acc = bp1[k];
#pragma unroll
        for (int c = 0; c < 9; ++c) acc += sDE[e*9 + c] * wp1[c*128 + k];
        ph[x] = fmaxf(acc, 0.0f);
    }
    __syncthreads();
    // pos MLP layer 2: [32,128]@[128,64]
    for (int x = tid; x < 32 * 64; x += 256) {
        int e = x >> 6, d = x & 63;
        float acc = bp2[d];
        for (int k = 0; k < 128; ++k) acc += ph[e*128 + k] * wp2[k*64 + d];
        sPos[x] = acc;
    }
    __syncthreads();                         // ph consumed; scratch free

    // pos half of edge input (head-invariant): one 16B LDS store per thread
    {
        const int e = tid >> 3;
        const int d0 = (tid & 7) * 8;
        v8h pv;
#pragma unroll
        for (int j = 0; j < 8; ++j) pv[j] = (half_t)sPos[e*64 + d0 + j];
        *(v8h*)&sX[e*128 + 64 + d0] = pv;
    }

    // Cache W1 B-fragments in registers: wave handles ntiles {2w, 2w+1}
    const int nt0 = wave * 2;
    v16h bf[2][4];
#pragma unroll
    for (int nti = 0; nti < 2; ++nti)
#pragma unroll
        for (int kc = 0; kc < 4; ++kc)
            bf[nti][kc] = load_frag(we1t, (nt0 + nti) * 16, kc * 32, 128);
    // Cache W2 B-fragments (used by waves 0,1 in GEMM2)
    v16h bw[8];
    if (wave < 2) {
#pragma unroll
        for (int kc = 0; kc < 8; ++kc) bw[kc] = load_frag(we2t, 0, kc * 32, 256);
    }

    for (int h = 0; h < HH; ++h) {
        if (tid < 64) sQ[tid] = qkv[(size_t)bi * 768 + h * 64 + tid];
        __syncthreads();
        // edge input first half: q_i * k_nb  (two b128 global loads + one 16B LDS store)
        {
            const int e = tid >> 3;
            const int d0 = (tid & 7) * 8;
            const float* krow = qkv + (size_t)(brow + sIdx[e]) * 768 + 256 + h * 64 + d0;
            v4f k0 = *(const v4f*)krow;
            v4f k1 = *(const v4f*)(krow + 4);
            v8h xv;
#pragma unroll
            for (int j = 0; j < 4; ++j) xv[j]     = (half_t)(sQ[d0 + j]     * k0[j]);
#pragma unroll
            for (int j = 0; j < 4; ++j) xv[4 + j] = (half_t)(sQ[d0 + 4 + j] * k1[j]);
            *(v8h*)&sX[e*128 + d0] = xv;
        }
        __syncthreads();

        // GEMM1: Hid = relu(X[32,128] @ W1[128,256] + b1) -> f16 scratch
        half_t* sHid = (half_t*)sScratch;
        v16h af[2][4];
#pragma unroll
        for (int mt = 0; mt < 2; ++mt)
#pragma unroll
            for (int kc = 0; kc < 4; ++kc)
                af[mt][kc] = load_frag(sX, mt * 16, kc * 32, 128);
#pragma unroll
        for (int nti = 0; nti < 2; ++nti) {
#pragma unroll
            for (int mt = 0; mt < 2; ++mt) {
                v8f acc = {};
#pragma unroll
                for (int kc = 0; kc < 4; ++kc)
                    acc = wmma_f16(af[mt][kc], bf[nti][kc], acc);
                int row = mt * 16 + ((lane >> 4) << 3);
                int col = (nt0 + nti) * 16 + (lane & 15);
                float bv = be1[col];
#pragma unroll
                for (int r = 0; r < 8; ++r)
                    sHid[(row + r) * 256 + col] = (half_t)fmaxf(acc[r] + bv, 0.0f);
            }
        }
        __syncthreads();

        // GEMM2: m = relu(Hid[32,256] @ W2[256,16] + b2) — waves 0,1
        if (wave < 2) {
            const int mt = wave;
            v8f acc = {};
#pragma unroll
            for (int kc = 0; kc < 8; ++kc) {
                v16h a = load_frag(sHid, mt * 16, kc * 32, 256);
                acc = wmma_f16(a, bw[kc], acc);
            }
            int row = mt * 16 + ((lane >> 4) << 3);
            int col = lane & 15;
            float bv = be2[col];
#pragma unroll
            for (int r = 0; r < 8; ++r)
                sM[(row + r) * 16 + col] = fmaxf(acc[r] + bv, 0.0f);
        }
        __syncthreads();

        // coor & attn head hiddens: [32,16]@[16,64] + relu (VALU, K tiny)
        float* sT = (float*)sScratch;   // [32*64] coor, then [32*64] attn
        {
            int e = tid >> 3;
            int k0 = (tid & 7) * 8;
#pragma unroll
            for (int kk = 0; kk < 8; ++kk) {
                int k = k0 + kk;
                float a1 = bc1[k], a2 = ba1[k];
#pragma unroll
                for (int j = 0; j < MDIM; ++j) {
                    float mv = sM[e*16 + j];
                    a1 += mv * wc1[j*64 + k];
                    a2 += mv * wa1[j*64 + k];
                }
                sT[e*64 + k]        = fmaxf(a1, 0.0f);
                sT[2048 + e*64 + k] = fmaxf(a2, 0.0f);
            }
        }
        __syncthreads();
        if (tid < 32) {
            float c2 = bc2[0], a2 = ba2[0];
#pragma unroll
            for (int k = 0; k < 64; ++k) {
                c2 += sT[tid*64 + k] * wc2[k];
                a2 += sT[2048 + tid*64 + k] * wa2[k];
            }
            sCW[tid]  = c2;
            sSim[tid] = a2;
        }
        __syncthreads();
        if (tid == 0) {   // softmax over 32 neighbors
            float mx = -3.4e38f;
            for (int e = 0; e < 32; ++e) mx = fmaxf(mx, sSim[e]);
            float s = 0.0f;
            for (int e = 0; e < 32; ++e) { float ev = expf(sSim[e] - mx); sAttn[e] = ev; s += ev; }
            float inv = 1.0f / s;
            for (int e = 0; e < 32; ++e) sAttn[e] *= inv;
        }
        if (tid < 3) {    // per-head coordinate contribution
            float acc = 0.0f;
            for (int e = 0; e < 32; ++e) acc += sCW[e] * sRel[e*3 + tid];
            sCAcc[tid] += wco[h] * acc;
        }
        __syncthreads();
        // attention output: sum_e attn[e] * (v_nb + pos_emb)
        if (tid < 64) {
            const int d = tid;
            float acc = 0.0f;
#pragma unroll 8
            for (int e = 0; e < 32; ++e) {
                float vv = qkv[(size_t)(brow + sIdx[e]) * 768 + 512 + h * 64 + d];
                acc += sAttn[e] * (vv + sPos[e*64 + d]);
            }
            outw[(size_t)bi * 256 + h * 64 + d] = acc;
        }
        __syncthreads();   // protect sX / scratch reuse next head
    }
    if (tid < 3) coors_out[(size_t)bi * 3 + tid] = sCAcc[tid];
}

// ---------------------------------------------------------------------------
// Host launcher
// ---------------------------------------------------------------------------
extern "C" void kernel_launch(void* const* d_in, const int* in_sizes, int n_in,
                              void* d_out, int out_size, void* d_ws, size_t ws_size,
                              hipStream_t stream) {
    (void)in_sizes; (void)n_in; (void)out_size; (void)ws_size;
    const float* feats  = (const float*)d_in[0];
    const float* coors  = (const float*)d_in[1];
    const float* w_qkv  = (const float*)d_in[2];
    const float* w_out  = (const float*)d_in[3];
    const float* b_out  = (const float*)d_in[4];
    const float* w_pos1 = (const float*)d_in[5];
    const float* b_pos1 = (const float*)d_in[6];
    const float* w_pos2 = (const float*)d_in[7];
    const float* b_pos2 = (const float*)d_in[8];
    const float* w_e1   = (const float*)d_in[9];
    const float* b_e1   = (const float*)d_in[10];
    const float* w_e2   = (const float*)d_in[11];
    const float* b_e2   = (const float*)d_in[12];
    const float* w_a1   = (const float*)d_in[13];
    const float* b_a1   = (const float*)d_in[14];
    const float* w_a2   = (const float*)d_in[15];
    const float* b_a2   = (const float*)d_in[16];
    const float* w_c1   = (const float*)d_in[17];
    const float* b_c1   = (const float*)d_in[18];
    const float* w_c2   = (const float*)d_in[19];
    const float* b_c2   = (const float*)d_in[20];
    const float* w_co   = (const float*)d_in[21];

    char* ws = (char*)d_ws;
    int*    ws_idx  = (int*)   (ws + OFF_IDX);
    float*  ws_nbd  = (float*) (ws + OFF_NBD);
    float*  ws_rel  = (float*) (ws + OFF_REL);
    float*  ws_qkv  = (float*) (ws + OFF_QKV);
    float*  ws_outw = (float*) (ws + OFF_OUTW);
    half_t* wqkvt   = (half_t*)(ws + OFF_WQKVT);
    half_t* woutt   = (half_t*)(ws + OFF_WOUTT);
    half_t* we1t    = (half_t*)(ws + OFF_WE1T);
    half_t* we2t    = (half_t*)(ws + OFF_WE2T);

    float* out_feats = (float*)d_out;                    // [BN,256]
    float* out_coors = (float*)d_out + (size_t)BN * DIM; // [BN,3]

    // 1. neighbors
    topk_kernel<<<BN, 256, 0, stream>>>(coors, ws_idx, ws_nbd, ws_rel);
    // 2. weight prep (f16, N-major)
    transpose_to_f16<<<(256*768 + 255)/256, 256, 0, stream>>>(w_qkv, wqkvt, 256, 768);
    transpose_to_f16<<<(256*256 + 255)/256, 256, 0, stream>>>(w_out, woutt, 256, 256);
    transpose_to_f16<<<(128*256 + 255)/256, 256, 0, stream>>>(w_e1,  we1t,  128, 256);
    transpose_to_f16<<<(256*16  + 255)/256, 256, 0, stream>>>(w_e2,  we2t,  256, 16);
    // 3. qkv = feats @ w_qkv (WMMA)
    gemm_wmma<768, 256><<<dim3(BN/64, 768/64), 256, 0, stream>>>(feats, wqkvt, nullptr, ws_qkv, BN);
    // 4. fused edge MLP (WMMA) + attention + coordinates
    fused_edge_attn<<<BN, 256, 0, stream>>>(
        ws_qkv, ws_idx, ws_nbd, ws_rel,
        we1t, b_e1, we2t, b_e2,
        w_pos1, b_pos1, w_pos2, b_pos2,
        w_a1, b_a1, w_a2, b_a2,
        w_c1, b_c1, w_c2, b_c2,
        w_co, ws_outw, out_coors);
    // 5. output projection (WMMA, with bias)
    gemm_wmma<256, 256><<<dim3(BN/64, 256/64), 256, 0, stream>>>(ws_outw, woutt, b_out, out_feats, BN);
}
